// Net_69217692942599
// MI455X (gfx1250) — compile-verified, matrix operand order checked by reading, not run on hardware
//
#include <hip/hip_runtime.h>
#include <math.h>

// ---------------------------------------------------------------------------
// CDNA5 (gfx1250) NNConv/GRU/Set2Set GNN forward pass.
// All dense matmuls use v_wmma_f32_16x16x32_f16 with pre-packed f16 operands
// (K padded to a multiple of 32) so the inner loop is pure b128 loads + WMMA —
// no predication, no per-element waits. The per-edge weight tensor ew
// [E,64,64] stays f16 end-to-end to halve the dominant HBM traffic.
// ---------------------------------------------------------------------------

typedef __attribute__((ext_vector_type(16))) _Float16 v16h;
typedef __attribute__((ext_vector_type(8)))  _Float16 v8h;
typedef __attribute__((ext_vector_type(2)))  _Float16 v2h;
typedef __attribute__((ext_vector_type(8)))  float    v8f;

union V16H { v16h v; struct { v8h lo; v8h hi; } p; _Float16 h[16]; };
union V8F  { v8f  v; float f[8]; };

__device__ __forceinline__ float sigm(float x) { return 1.0f / (1.0f + __expf(-x)); }

// monotone float <-> uint mapping for atomic max on floats
__device__ __forceinline__ unsigned f2mono(float f) {
  unsigned u = __float_as_uint(f);
  return (u & 0x80000000u) ? ~u : (u | 0x80000000u);
}
__device__ __forceinline__ float mono2f(unsigned m) {
  unsigned u = (m & 0x80000000u) ? (m ^ 0x80000000u) : ~m;
  return __uint_as_float(u);
}

// ---------------------------------------------------------------------------
// Pack f32 [M,K] -> f16 [M,Kp] (zero-padded columns K..Kp-1).
// ---------------------------------------------------------------------------
__global__ void pack_f16(const float* __restrict__ X, _Float16* __restrict__ Y,
                         long M, int K, int Kp)
{
  long i = (long)blockIdx.x * blockDim.x + threadIdx.x;
  if (i >= M * (long)Kp) return;
  long m = i / Kp;
  int  k = (int)(i - m * Kp);
  Y[i] = (_Float16)((k < K) ? X[m * (long)K + k] : 0.0f);
}

// ---------------------------------------------------------------------------
// GEMM: C[M,Nc] = Ah[M,Kp] * Wh[Nc,Kp]^T + bias[Nc]   (f16 in, f32 acc)
// One 16x16 tile per wave, 4 waves per block. Fragment layout (ISA 7.12.2):
//   lane l -> m/n = l&15, kb = (l>>4)*8 ; elems j=0..7 -> k=kb+j (contig 16B),
//   j=8..15 -> k=kb+16+(j-8) (contig 16B)  => two b128 loads per matrix/step.
// OOB rows are clamped to row 0 (never stored) — no divergence, EXEC all-1s.
// ---------------------------------------------------------------------------
template <typename OutT>
__global__ __launch_bounds__(128)
void gemm_wmma(const _Float16* __restrict__ A, const _Float16* __restrict__ W,
               const float* __restrict__ bias, OutT* __restrict__ C,
               int M, int Kp, int Nc)
{
  const int lane = threadIdx.x & 31;
  const int wave = threadIdx.x >> 5;
  const int tn   = blockIdx.x * 16;
  const int tm   = blockIdx.y * 64 + wave * 16;
  int mA = tm + (lane & 15); if (mA >= M)  mA = 0;
  int nB = tn + (lane & 15); if (nB >= Nc) nB = 0;
  const int kb = (lane >> 4) * 8;
  const _Float16* __restrict__ Ap = A + (size_t)mA * Kp + kb;
  const _Float16* __restrict__ Wp = W + (size_t)nB * Kp + kb;

  v8f acc = {0.f, 0.f, 0.f, 0.f, 0.f, 0.f, 0.f, 0.f};

#define WMMA_STEP(K0)                                                      \
  {                                                                        \
    V16H a_, b_;                                                           \
    a_.p.lo = *(const v8h*)(Ap + (K0));                                    \
    a_.p.hi = *(const v8h*)(Ap + (K0) + 16);                               \
    b_.p.lo = *(const v8h*)(Wp + (K0));                                    \
    b_.p.hi = *(const v8h*)(Wp + (K0) + 16);                               \
    acc = __builtin_amdgcn_wmma_f32_16x16x32_f16(false, a_.v, false, b_.v, \
                                                 (short)0, acc, false, false); \
  }

  int k0 = 0;
  for (; k0 + 64 <= Kp; k0 += 64) {
    if (k0 + 64 < Kp) {
      __builtin_prefetch(Ap + k0 + 64, 0, 0);
      __builtin_prefetch(Wp + k0 + 64, 0, 0);
    }
    WMMA_STEP(k0)
    WMMA_STEP(k0 + 32)
  }
  if (k0 < Kp) WMMA_STEP(k0)
#undef WMMA_STEP

  V8F r; r.v = acc;
  const int cn = tn + (lane & 15);
  const int rb = tm + 8 * (lane >> 4);
  if (cn < Nc) {
    float bv = bias[cn];
#pragma unroll
    for (int v = 0; v < 8; ++v) {
      int cm = rb + v;
      if (cm < M) C[(size_t)cm * Nc + cn] = (OutT)(r.f[v] + bv);
    }
  }
}

// ---------------------------------------------------------------------------
// BatchNorm: per-column batch stats (biased var) folded into scale/shift.
// blockDim = (32,8), gridDim.x = C/32 (C multiple of 32 here: 64, 4096).
// ---------------------------------------------------------------------------
template <typename T>
__global__ void bn_stats(const T* __restrict__ Y, int M, int C,
                         const float* __restrict__ g, const float* __restrict__ be,
                         float* __restrict__ scale, float* __restrict__ shift)
{
  int c = blockIdx.x * 32 + threadIdx.x;
  float s = 0.f, ss = 0.f;
  for (int r = threadIdx.y; r < M; r += 8) {
    float v = (float)Y[(size_t)r * C + c];
    s += v; ss += v * v;
  }
  __shared__ float Sb[8][32], SSb[8][32];
  Sb[threadIdx.y][threadIdx.x] = s;
  SSb[threadIdx.y][threadIdx.x] = ss;
  __syncthreads();
  if (threadIdx.y == 0) {
#pragma unroll
    for (int i = 1; i < 8; ++i) { s += Sb[i][threadIdx.x]; ss += SSb[i][threadIdx.x]; }
    float mean = s / (float)M;
    float var  = ss / (float)M - mean * mean;
    float inv  = rsqrtf(var + 1e-5f);
    float sc   = g[c] * inv;
    scale[c] = sc;
    shift[c] = be[c] - mean * sc;
  }
}

template <typename T>
__global__ void bn_apply(T* __restrict__ Y, const float* __restrict__ scale,
                         const float* __restrict__ shift, long total, int C, int relu)
{
  long i = (long)blockIdx.x * blockDim.x + threadIdx.x;
  if (i >= total) return;
  int c = (int)(i % C);
  float v = (float)Y[i] * scale[c] + shift[c];
  if (relu) v = fmaxf(v, 0.f);
  Y[i] = (T)v;
}

// ---------------------------------------------------------------------------
// LayerNorm over last dim; one wave32 per row, shuffle reduction. In-place OK.
// ---------------------------------------------------------------------------
__global__ void ln_rows(const float* __restrict__ X, const float* __restrict__ g,
                        const float* __restrict__ b, float* __restrict__ Yo,
                        int M, int Dd, int relu)
{
  int row  = blockIdx.x * (blockDim.x >> 5) + (threadIdx.x >> 5);
  int lane = threadIdx.x & 31;
  if (row >= M) return;
  const float* xr = X + (size_t)row * Dd;
  float s = 0.f, ss = 0.f;
  for (int j = lane; j < Dd; j += 32) { float v = xr[j]; s += v; ss += v * v; }
#pragma unroll
  for (int off = 16; off; off >>= 1) {
    s  += __shfl_xor(s,  off, 32);
    ss += __shfl_xor(ss, off, 32);
  }
  float mean = s / (float)Dd;
  float var  = ss / (float)Dd - mean * mean;
  float inv  = rsqrtf(var + 1e-5f);
  float* yr = Yo + (size_t)row * Dd;
  for (int j = lane; j < Dd; j += 32) {
    float v = (xr[j] - mean) * inv * g[j] + b[j];
    if (relu) v = fmaxf(v, 0.f);
    yr[j] = v;
  }
}

// ---------------------------------------------------------------------------
__global__ void fill_f32(float* __restrict__ p, float v, long n) {
  long i = (long)blockIdx.x * blockDim.x + threadIdx.x;
  if (i < n) p[i] = v;
}
__global__ void fill_u32(unsigned* __restrict__ p, unsigned v, long n) {
  long i = (long)blockIdx.x * blockDim.x + threadIdx.x;
  if (i < n) p[i] = v;
}

__global__ void deg_count(const int* __restrict__ dst, float* __restrict__ deg, int E) {
  int e = blockIdx.x * blockDim.x + threadIdx.x;
  if (e < E) atomicAdd(&deg[dst[e]], 1.0f);
}

// Per-edge 64x64 matvec msg = x_src @ ew[e]; scatter-add into agg[dst].
// One wave per edge; lane handles columns (2*lane, 2*lane+1) -> one dword
// f16x2 load per row per lane, fully coalesced.
__global__ __launch_bounds__(256)
void nnconv_msg(const float* __restrict__ out, const _Float16* __restrict__ ew,
                const int* __restrict__ src, const int* __restrict__ dst,
                float* __restrict__ agg, int E)
{
  int e    = blockIdx.x * 8 + (threadIdx.x >> 5);
  int lane = threadIdx.x & 31;
  if (e >= E) return;
  int sn = src[e], dn = dst[e];
  const float* xs = out + (size_t)sn * 64;
  const v2h* We2 = (const v2h*)(ew + (size_t)e * 4096);
  float a0 = 0.f, a1 = 0.f;
#pragma unroll 4
  for (int i = 0; i < 64; ++i) {
    float xi = xs[i];
    v2h w = We2[i * 32 + lane];
    a0 += xi * (float)w.x;
    a1 += xi * (float)w.y;
  }
  atomicAdd(&agg[(size_t)dn * 64 + 2 * lane],     a0);
  atomicAdd(&agg[(size_t)dn * 64 + 2 * lane + 1], a1);
}

__global__ void agg_finalize(const float* __restrict__ agg, const float* __restrict__ deg,
                             const float* __restrict__ cbias, float* __restrict__ m, int N)
{
  int i = blockIdx.x * blockDim.x + threadIdx.x;
  if (i >= N * 64) return;
  int n = i >> 6, c = i & 63;
  float d = fmaxf(deg[n], 1.0f);
  m[i] = fmaxf(agg[i] / d + cbias[c], 0.f);
}

__global__ void gru_cell(const float* __restrict__ gi, const float* __restrict__ gh,
                         float* __restrict__ h, int N)
{
  int i = blockIdx.x * blockDim.x + threadIdx.x;
  if (i >= N * 64) return;
  int n = i >> 6, c = i & 63;
  const float* gir = gi + (size_t)n * 192;
  const float* ghr = gh + (size_t)n * 192;
  float r  = sigm(gir[c]        + ghr[c]);
  float z  = sigm(gir[64 + c]   + ghr[64 + c]);
  float nn = tanhf(gir[128 + c] + r * ghr[128 + c]);
  h[i] = (1.f - z) * nn + z * h[i];
}

__global__ void lstm_cell(const float* __restrict__ g1, const float* __restrict__ g2,
                          float* __restrict__ cs, float* __restrict__ hs, int B)
{
  int i = blockIdx.x * blockDim.x + threadIdx.x;
  if (i >= B * 64) return;
  int b = i >> 6, c = i & 63;
  const float* a = g1 + (size_t)b * 256;
  const float* d = g2 + (size_t)b * 256;
  float gi_ = a[c]       + d[c];
  float gf_ = a[64 + c]  + d[64 + c];
  float gg_ = a[128 + c] + d[128 + c];
  float go_ = a[192 + c] + d[192 + c];
  float cn = sigm(gf_) * cs[i] + sigm(gi_) * tanhf(gg_);
  cs[i] = cn;
  hs[i] = sigm(go_) * tanhf(cn);
}

// attention: e[n] = dot(out[n,:], hs[batch[n],:]); track segment max
__global__ void attn_dot(const float* __restrict__ out, const float* __restrict__ hs,
                         const int* __restrict__ batch, float* __restrict__ e_n,
                         unsigned* __restrict__ emaxU, int N)
{
  int n    = blockIdx.x * (blockDim.x >> 5) + (threadIdx.x >> 5);
  int lane = threadIdx.x & 31;
  if (n >= N) return;
  int b = batch[n];
  const float* o = out + (size_t)n * 64;
  const float* q = hs  + (size_t)b * 64;
  float s = o[lane] * q[lane] + o[lane + 32] * q[lane + 32];
#pragma unroll
  for (int off = 16; off; off >>= 1) s += __shfl_xor(s, off, 32);
  if (lane == 0) {
    e_n[n] = s;
    atomicMax(&emaxU[b], f2mono(s));
  }
}

__global__ void attn_exp(const float* __restrict__ e_n, const unsigned* __restrict__ emaxU,
                         const int* __restrict__ batch, float* __restrict__ ee,
                         float* __restrict__ esum, int N)
{
  int n = blockIdx.x * blockDim.x + threadIdx.x;
  if (n >= N) return;
  int b = batch[n];
  float v = __expf(e_n[n] - mono2f(emaxU[b]));
  ee[n] = v;
  atomicAdd(&esum[b], v);
}

__global__ void attn_scatter(const float* __restrict__ ee, const float* __restrict__ esum,
                             const float* __restrict__ out, const int* __restrict__ batch,
                             float* __restrict__ rvec, int N)
{
  int i = blockIdx.x * blockDim.x + threadIdx.x;
  if (i >= N * 64) return;
  int n = i >> 6, c = i & 63;
  int b = batch[n];
  float a = ee[n] / fmaxf(esum[b], 1e-16f);
  atomicAdd(&rvec[(size_t)b * 64 + c], a * out[i]);
}

__global__ void build_qstar(const float* __restrict__ hs, const float* __restrict__ rvec,
                            float* __restrict__ qstar, int B)
{
  int i = blockIdx.x * blockDim.x + threadIdx.x;
  if (i >= B * 128) return;
  int b = i >> 7, c = i & 127;
  qstar[i] = (c < 64) ? hs[(size_t)b * 64 + c] : rvec[(size_t)b * 64 + (c - 64)];
}

__global__ void build_zz(const float* __restrict__ out, const float* __restrict__ qstar,
                         const int* __restrict__ ti0, const int* __restrict__ ti1,
                         const int* __restrict__ tb, float* __restrict__ zz, int T)
{
  long i = (long)blockIdx.x * blockDim.x + threadIdx.x;
  if (i >= (long)T * 256) return;
  int t = (int)(i >> 8), c = (int)(i & 255);
  float v;
  if (c < 64)       v = out[(size_t)ti0[t] * 64 + c];
  else if (c < 128) v = out[(size_t)ti1[t] * 64 + (c - 64)];
  else              v = qstar[(size_t)tb[t] * 128 + (c - 128)];
  zz[i] = v;
}

__global__ void gather_out(const float* __restrict__ logits, const int* __restrict__ cls,
                           float* __restrict__ o, int T)
{
  int t = blockIdx.x * blockDim.x + threadIdx.x;
  if (t < T) o[t] = logits[(size_t)t * 8 + cls[t]];
}

// ---------------------------------------------------------------------------
// Host orchestration
// ---------------------------------------------------------------------------
static inline long cdivl(long a, long b) { return (a + b - 1) / b; }

extern "C" void kernel_launch(void* const* d_in, const int* in_sizes, int n_in,
                              void* d_out, int out_size, void* d_ws, size_t ws_size,
                              hipStream_t stream)
{
  const int N = 12000, E = 60000, B = 600, T = 24000;
  (void)n_in; (void)in_sizes; (void)out_size; (void)ws_size;

  // inputs
  const float* x         = (const float*)d_in[0];
  const float* edge_attr = (const float*)d_in[1];
  const int*   src       = (const int*)d_in[2];
  const int*   dstp      = ((const int*)d_in[2]) + E;
  const int*   batch     = (const int*)d_in[3];
  const int*   ti0       = (const int*)d_in[4];
  const int*   ti1       = ((const int*)d_in[4]) + T;
  const int*   tb        = (const int*)d_in[5];
  const int*   tcls      = (const int*)d_in[6];
  // params (setup_inputs dict order)
  const float* pre_W1 = (const float*)d_in[7];
  const float* pre_b1 = (const float*)d_in[8];
  const float* pre_g1 = (const float*)d_in[9];
  const float* pre_be1= (const float*)d_in[10];
  const float* pre_W2 = (const float*)d_in[11];
  const float* pre_b2 = (const float*)d_in[12];
  const float* pre_g2 = (const float*)d_in[13];
  const float* pre_be2= (const float*)d_in[14];
  const float* enc_W1 = (const float*)d_in[15];
  const float* enc_b1 = (const float*)d_in[16];
  const float* enc_g1 = (const float*)d_in[17];
  const float* enc_be1= (const float*)d_in[18];
  const float* enc_W2 = (const float*)d_in[19];
  const float* enc_b2 = (const float*)d_in[20];
  const float* enc_g2 = (const float*)d_in[21];
  const float* enc_be2= (const float*)d_in[22];
  const float* conv_b = (const float*)d_in[23];
  const float* gWih   = (const float*)d_in[24];
  const float* gWhh   = (const float*)d_in[25];
  const float* gbih   = (const float*)d_in[26];
  const float* gbhh   = (const float*)d_in[27];
  const float* glng   = (const float*)d_in[28];
  const float* glnb   = (const float*)d_in[29];
  const float* lWih   = (const float*)d_in[30];
  const float* lWhh   = (const float*)d_in[31];
  const float* lbih   = (const float*)d_in[32];
  const float* lbhh   = (const float*)d_in[33];
  const float* pW1    = (const float*)d_in[34];
  const float* pb1    = (const float*)d_in[35];
  const float* pg1    = (const float*)d_in[36];
  const float* pbe1   = (const float*)d_in[37];
  const float* pW2    = (const float*)d_in[38];
  const float* pb2    = (const float*)d_in[39];
  const float* pg2    = (const float*)d_in[40];
  const float* pbe2   = (const float*)d_in[41];
  const float* pW3    = (const float*)d_in[42];
  const float* pb3    = (const float*)d_in[43];

  // workspace carving (256B aligned)
  char* wp = (char*)d_ws;
  auto carve = [&](size_t bytes) -> char* {
    char* r = wp;
    wp += (bytes + 255) & ~(size_t)255;
    return r;
  };
  // big f16 buffers
  _Float16* ew    = (_Float16*)carve((size_t)E * 4096 * sizeof(_Float16));
  // f32 activations
  float* e1    = (float*)carve((size_t)E * 64 * sizeof(float));
  float* ybuf  = (float*)carve((size_t)N * 64 * sizeof(float));
  float* outb  = (float*)carve((size_t)N * 64 * sizeof(float));
  float* h     = (float*)carve((size_t)N * 64 * sizeof(float));
  float* mbuf  = (float*)carve((size_t)N * 64 * sizeof(float));
  float* agg   = (float*)carve((size_t)N * 64 * sizeof(float));
  float* gi    = (float*)carve((size_t)N * 192 * sizeof(float));
  float* gh    = (float*)carve((size_t)N * 192 * sizeof(float));
  float* deg   = (float*)carve((size_t)N * sizeof(float));
  float* scale = (float*)carve((size_t)4096 * sizeof(float));
  float* shift = (float*)carve((size_t)4096 * sizeof(float));
  float* qstar = (float*)carve((size_t)B * 128 * sizeof(float));
  float* hsb   = (float*)carve((size_t)B * 64 * sizeof(float));
  float* csb   = (float*)carve((size_t)B * 64 * sizeof(float));
  float* g1b   = (float*)carve((size_t)B * 256 * sizeof(float));
  float* g2b   = (float*)carve((size_t)B * 256 * sizeof(float));
  float* e_n   = (float*)carve((size_t)N * sizeof(float));
  float* eeb   = (float*)carve((size_t)N * sizeof(float));
  unsigned* emaxU = (unsigned*)carve((size_t)B * sizeof(unsigned));
  float* esum  = (float*)carve((size_t)B * sizeof(float));
  float* rvec  = (float*)carve((size_t)B * 64 * sizeof(float));
  float* zz    = (float*)carve((size_t)T * 256 * sizeof(float));
  float* z2    = (float*)carve((size_t)T * 256 * sizeof(float));
  float* logits= (float*)carve((size_t)T * 8 * sizeof(float));
  // packed f16 weights (rows x Kp)
  _Float16* pre_W1h = (_Float16*)carve((size_t)64 * 32 * 2);
  _Float16* pre_W2h = (_Float16*)carve((size_t)64 * 64 * 2);
  _Float16* enc_W1h = (_Float16*)carve((size_t)64 * 32 * 2);
  _Float16* enc_W2h = (_Float16*)carve((size_t)4096 * 64 * 2);
  _Float16* gWihh   = (_Float16*)carve((size_t)192 * 64 * 2);
  _Float16* gWhhh   = (_Float16*)carve((size_t)192 * 64 * 2);
  _Float16* lWihh   = (_Float16*)carve((size_t)256 * 128 * 2);
  _Float16* lWhhh   = (_Float16*)carve((size_t)256 * 64 * 2);
  _Float16* pW1h    = (_Float16*)carve((size_t)256 * 256 * 2);
  _Float16* pW2h    = (_Float16*)carve((size_t)256 * 256 * 2);
  _Float16* pW3h    = (_Float16*)carve((size_t)8 * 256 * 2);
  // packed f16 activations
  _Float16* xh    = (_Float16*)carve((size_t)N * 32 * 2);
  _Float16* eah   = (_Float16*)carve((size_t)E * 32 * 2);
  _Float16* act64 = (_Float16*)carve((size_t)E * 64 * 2);  // N*64 and E*64 reuse
  _Float16* hh    = (_Float16*)carve((size_t)N * 64 * 2);
  _Float16* qh    = (_Float16*)carve((size_t)B * 128 * 2);
  _Float16* hsh   = (_Float16*)carve((size_t)B * 64 * 2);
  _Float16* zzh   = (_Float16*)carve((size_t)T * 256 * 2);

  auto pack = [&](const float* X, _Float16* Y, long M, int K, int Kp) {
    long total = M * (long)Kp;
    pack_f16<<<(unsigned)cdivl(total, 256), 256, 0, stream>>>(X, Y, M, K, Kp);
  };
  auto gemmF = [&](const _Float16* A, const _Float16* W, const float* bias, float* C,
                   int M, int Kp, int Nc) {
    dim3 g((Nc + 15) / 16, (M + 63) / 64);
    gemm_wmma<float><<<g, 128, 0, stream>>>(A, W, bias, C, M, Kp, Nc);
  };
  auto gemmH = [&](const _Float16* A, const _Float16* W, const float* bias, _Float16* C,
                   int M, int Kp, int Nc) {
    dim3 g((Nc + 15) / 16, (M + 63) / 64);
    gemm_wmma<_Float16><<<g, 128, 0, stream>>>(A, W, bias, C, M, Kp, Nc);
  };
  auto bnF = [&](float* Y, int M, int C, const float* g, const float* be, int relu) {
    bn_stats<float><<<dim3(C / 32), dim3(32, 8), 0, stream>>>(Y, M, C, g, be, scale, shift);
    long total = (long)M * C;
    bn_apply<float><<<(unsigned)cdivl(total, 256), 256, 0, stream>>>(Y, scale, shift, total, C, relu);
  };

  // ---- pack all weights once
  pack(pre_W1, pre_W1h, 64, 16, 32);
  pack(pre_W2, pre_W2h, 64, 64, 64);
  pack(enc_W1, enc_W1h, 64, 16, 32);
  pack(enc_W2, enc_W2h, 4096, 64, 64);
  pack(gWih, gWihh, 192, 64, 64);
  pack(gWhh, gWhhh, 192, 64, 64);
  pack(lWih, lWihh, 256, 128, 128);
  pack(lWhh, lWhhh, 256, 64, 64);
  pack(pW1, pW1h, 256, 256, 256);
  pack(pW2, pW2h, 256, 256, 256);
  pack(pW3, pW3h, 8, 256, 256);

  // ---- preprocess MLP: Linear-BN-ReLU x2 -> outb; h = outb
  pack(x, xh, N, 16, 32);
  gemmF(xh, pre_W1h, pre_b1, ybuf, N, 32, 64);
  bnF(ybuf, N, 64, pre_g1, pre_be1, 1);
  pack(ybuf, act64, N, 64, 64);
  gemmF(act64, pre_W2h, pre_b2, outb, N, 64, 64);
  bnF(outb, N, 64, pre_g2, pre_be2, 1);
  hipMemcpyAsync(h, outb, (size_t)N * 64 * sizeof(float), hipMemcpyDeviceToDevice, stream);

  // ---- edge network -> ew [E,4096] in f16
  pack(edge_attr, eah, E, 16, 32);
  gemmF(eah, enc_W1h, enc_b1, e1, E, 32, 64);
  bnF(e1, E, 64, enc_g1, enc_be1, 1);
  pack(e1, act64, E, 64, 64);
  gemmH(act64, enc_W2h, enc_b2, ew, E, 64, 4096);
  bn_stats<_Float16><<<dim3(4096 / 32), dim3(32, 8), 0, stream>>>(ew, E, 4096, enc_g2, enc_be2, scale, shift);
  {
    long total = (long)E * 4096;
    bn_apply<_Float16><<<(unsigned)cdivl(total, 256), 256, 0, stream>>>(ew, scale, shift, total, 4096, 0);
  }

  // ---- degree
  fill_f32<<<(unsigned)cdivl(N, 256), 256, 0, stream>>>(deg, 0.f, N);
  deg_count<<<(unsigned)cdivl(E, 256), 256, 0, stream>>>(dstp, deg, E);

  // ---- 3 message-passing + GRU steps
  for (int step = 0; step < 3; ++step) {
    fill_f32<<<(unsigned)cdivl((long)N * 64, 256), 256, 0, stream>>>(agg, 0.f, (long)N * 64);
    nnconv_msg<<<(unsigned)cdivl(E, 8), 256, 0, stream>>>(outb, ew, src, dstp, agg, E);
    agg_finalize<<<(unsigned)cdivl((long)N * 64, 256), 256, 0, stream>>>(agg, deg, conv_b, mbuf, N);
    pack(mbuf, act64, N, 64, 64);
    gemmF(act64, gWihh, gbih, gi, N, 64, 192);
    pack(h, hh, N, 64, 64);
    gemmF(hh, gWhhh, gbhh, gh, N, 64, 192);
    gru_cell<<<(unsigned)cdivl((long)N * 64, 256), 256, 0, stream>>>(gi, gh, h, N);
    ln_rows<<<(unsigned)cdivl(N, 8), 256, 0, stream>>>(h, glng, glnb, outb, N, 64, 0);
  }

  // ---- Set2Set (3 steps)
  fill_f32<<<(unsigned)cdivl((long)B * 128, 256), 256, 0, stream>>>(qstar, 0.f, (long)B * 128);
  fill_f32<<<(unsigned)cdivl((long)B * 64, 256), 256, 0, stream>>>(hsb, 0.f, (long)B * 64);
  fill_f32<<<(unsigned)cdivl((long)B * 64, 256), 256, 0, stream>>>(csb, 0.f, (long)B * 64);
  for (int step = 0; step < 3; ++step) {
    pack(qstar, qh, B, 128, 128);
    gemmF(qh, lWihh, lbih, g1b, B, 128, 256);
    pack(hsb, hsh, B, 64, 64);
    gemmF(hsh, lWhhh, lbhh, g2b, B, 64, 256);
    lstm_cell<<<(unsigned)cdivl((long)B * 64, 256), 256, 0, stream>>>(g1b, g2b, csb, hsb, B);
    fill_u32<<<(unsigned)cdivl(B, 256), 256, 0, stream>>>(emaxU, 0x007FFFFFu, B); // mono(-inf)
    fill_f32<<<(unsigned)cdivl(B, 256), 256, 0, stream>>>(esum, 0.f, B);
    fill_f32<<<(unsigned)cdivl((long)B * 64, 256), 256, 0, stream>>>(rvec, 0.f, (long)B * 64);
    attn_dot<<<(unsigned)cdivl(N, 8), 256, 0, stream>>>(outb, hsb, batch, e_n, emaxU, N);
    attn_exp<<<(unsigned)cdivl(N, 256), 256, 0, stream>>>(e_n, emaxU, batch, eeb, esum, N);
    attn_scatter<<<(unsigned)cdivl((long)N * 64, 256), 256, 0, stream>>>(eeb, esum, outb, batch, rvec, N);
    build_qstar<<<(unsigned)cdivl((long)B * 128, 256), 256, 0, stream>>>(hsb, rvec, qstar, B);
  }

  // ---- predictor head
  build_zz<<<(unsigned)cdivl((long)T * 256, 256), 256, 0, stream>>>(outb, qstar, ti0, ti1, tb, zz, T);
  pack(zz, zzh, T, 256, 256);
  gemmF(zzh, pW1h, pb1, z2, T, 256, 256);
  ln_rows<<<(unsigned)cdivl(T, 8), 256, 0, stream>>>(z2, pg1, pbe1, z2, T, 256, 1);
  pack(z2, zzh, T, 256, 256);
  gemmF(zzh, pW2h, pb2, zz, T, 256, 256);
  ln_rows<<<(unsigned)cdivl(T, 8), 256, 0, stream>>>(zz, pg2, pbe2, zz, T, 256, 1);
  pack(zz, zzh, T, 256, 256);
  gemmF(zzh, pW3h, pb3, logits, T, 256, 8);
  gather_out<<<(unsigned)cdivl(T, 256), 256, 0, stream>>>(logits, tcls, (float*)d_out, T);
}
